// ResidualBlock_89730456748195
// MI455X (gfx1250) — compile-verified, matrix operand order checked by reading, not run on hardware
//
#include <hip/hip_runtime.h>
#include <hip/hip_bf16.h>

// ---------------- problem constants ----------------
constexpr int D_MODEL  = 768;
constexpr int D_INNER  = 1536;
constexpr int N_STATE  = 16;
constexpr int D_CONV   = 4;
constexpr int DT_RANK  = 48;
constexpr int DT_PAD   = 64;               // dt_rank padded to multiple of 32 for WMMA K
constexpr int BB       = 2;
constexpr int LL       = 2048;
constexpr int NTOK     = BB * LL;          // 4096 tokens
constexpr int DBC_W    = DT_RANK + 2 * N_STATE;  // 80

typedef __attribute__((ext_vector_type(16))) _Float16 v16h;
typedef __attribute__((ext_vector_type(8)))  _Float16 v8h;
typedef __attribute__((ext_vector_type(8)))  float    v8f;

// ---------------- workspace layout (all chunks multiple of 256B) ----------------
constexpr size_t OFF_XN16   = 0;                                     // 4096*768 f16
constexpr size_t OFF_WIN16  = OFF_XN16   + (size_t)NTOK*D_MODEL*2;   // 3072*768 f16
constexpr size_t OFF_WOUT16 = OFF_WIN16  + (size_t)2*D_INNER*D_MODEL*2;
constexpr size_t OFF_WXP16  = OFF_WOUT16 + (size_t)D_MODEL*D_INNER*2;
constexpr size_t OFF_WDT16  = OFF_WXP16  + (size_t)DBC_W*D_INNER*2;  // 1536*64 f16 (padded)
constexpr size_t OFF_XIN32  = OFF_WDT16  + (size_t)D_INNER*DT_PAD*2; // 4096*1536 f32 (aliased as DELTA)
constexpr size_t OFF_RESG16 = OFF_XIN32  + (size_t)NTOK*D_INNER*4;   // silu(res) f16
constexpr size_t OFF_XC32   = OFF_RESG16 + (size_t)NTOK*D_INNER*2;   // conv+silu f32
constexpr size_t OFF_XC16   = OFF_XC32   + (size_t)NTOK*D_INNER*4;   // conv+silu f16
constexpr size_t OFF_DBC32  = OFF_XC16   + (size_t)NTOK*D_INNER*2;   // 4096*80 f32
constexpr size_t OFF_DT16   = OFF_DBC32  + (size_t)NTOK*DBC_W*4;     // 4096*64 f16 (padded)
constexpr size_t OFF_YG16   = OFF_DT16   + (size_t)NTOK*DT_PAD*2;    // gated scan out f16
constexpr size_t OFF_DELTA  = OFF_XIN32;                             // alias: xin dead after conv

// ---------------- layernorm -> f16 ----------------
__global__ __launch_bounds__(256)
void ln_kernel(const float* __restrict__ x, const float* __restrict__ w,
               const float* __restrict__ b, _Float16* __restrict__ xn)
{
    __shared__ float s1[256], s2[256];
    const int t = blockIdx.x;
    const float* xr = x + (size_t)t * D_MODEL;
    float sum = 0.f, sq = 0.f;
    for (int i = threadIdx.x; i < D_MODEL; i += 256) { float v = xr[i]; sum += v; sq += v * v; }
    s1[threadIdx.x] = sum; s2[threadIdx.x] = sq;
    __syncthreads();
    for (int s = 128; s > 0; s >>= 1) {
        if ((int)threadIdx.x < s) { s1[threadIdx.x] += s1[threadIdx.x + s]; s2[threadIdx.x] += s2[threadIdx.x + s]; }
        __syncthreads();
    }
    const float mu   = s1[0] * (1.f / D_MODEL);
    const float var  = s2[0] * (1.f / D_MODEL) - mu * mu;
    const float rinv = rsqrtf(var + 1e-5f);
    for (int i = threadIdx.x; i < D_MODEL; i += 256)
        xn[(size_t)t * D_MODEL + i] = (_Float16)((xr[i] - mu) * rinv * w[i] + b[i]);
}

// ---------------- f32 -> f16 converters ----------------
__global__ __launch_bounds__(256)
void cvt_kernel(const float* __restrict__ src, _Float16* __restrict__ dst, int n)
{
    int i = blockIdx.x * 256 + threadIdx.x;
    if (i < n) dst[i] = (_Float16)src[i];
}

__global__ __launch_bounds__(256)
void cvt_pad_kernel(const float* __restrict__ src, _Float16* __restrict__ dst,
                    int rows, int srcld, int dstld, int copyc)
{
    int i = blockIdx.x * 256 + threadIdx.x;
    if (i >= rows * dstld) return;
    int r = i / dstld, c = i - r * dstld;
    dst[i] = (c < copyc) ? (_Float16)src[(size_t)r * srcld + c] : (_Float16)0.f;
}

// ---------------- register-blocked WMMA f16->f32 GEMM: C = A(MxK) * B(NxK)^T ----------
// Each wave owns an MBLK x NBLK grid of 16x16 tiles: per k-step it loads
// MBLK A-fragments + NBLK B-fragments and issues MBLK*NBLK independent WMMAs
// (A reused NBLK times, B reused MBLK times, accumulators hide XDL latency).
// EPI 0: plain f32 store      (ldc = N)
// EPI 1: softplus(bias + v)   (delta epilogue, ldc = N)
// EPI 2: in_proj split: col < D_INNER -> f32 xin ; else silu -> f16 resg
template<int MBLK, int NBLK, int EPI>
__global__ __launch_bounds__(256)
void gemm_wmma(const _Float16* __restrict__ A, const _Float16* __restrict__ B,
               float* __restrict__ C, int M, int N, int K,
               const float* __restrict__ bias, _Float16* __restrict__ out2)
{
    const int lane    = threadIdx.x & 31;
    const int wave    = threadIdx.x >> 5;
    const int nTiles  = N >> 4;
    const int mBlocks = (M >> 4) / MBLK;
    const int nBlocks = nTiles / NBLK;
    const int blk     = blockIdx.x * 8 + wave;
    if (blk >= mBlocks * nBlocks) return;
    const int mB = blk / nBlocks;
    const int nB = blk - mB * nBlocks;
    const int mr = lane & 15;
    const int hi = lane >> 4;

    // CDNA5 16-bit fragment striping: lanes 0-15 row M, hi-half gets K+8 / K+24 groups.
    const _Float16* Ap[MBLK];
    const _Float16* Bp[NBLK];
    #pragma unroll
    for (int i = 0; i < MBLK; ++i)
        Ap[i] = A + (size_t)((mB * MBLK + i) * 16 + mr) * K + hi * 8;
    #pragma unroll
    for (int j = 0; j < NBLK; ++j)
        Bp[j] = B + (size_t)((nB * NBLK + j) * 16 + mr) * K + hi * 8;

    v8f acc[MBLK][NBLK];
    const v8f vzero = {};
    #pragma unroll
    for (int i = 0; i < MBLK; ++i)
        #pragma unroll
        for (int j = 0; j < NBLK; ++j) acc[i][j] = vzero;

    for (int k0 = 0; k0 < K; k0 += 32) {
        union Frag { v16h v; v8h h[2]; } a[MBLK], b[NBLK];
        #pragma unroll
        for (int i = 0; i < MBLK; ++i) {
            a[i].h[0] = *(const v8h*)(Ap[i] + k0);
            a[i].h[1] = *(const v8h*)(Ap[i] + k0 + 16);
        }
        #pragma unroll
        for (int j = 0; j < NBLK; ++j) {
            b[j].h[0] = *(const v8h*)(Bp[j] + k0);
            b[j].h[1] = *(const v8h*)(Bp[j] + k0 + 16);
        }
        if (k0 + 96 < K) {
            __builtin_prefetch((const void*)(Ap[0] + k0 + 96), 0, 3);
            __builtin_prefetch((const void*)(Bp[0] + k0 + 96), 0, 3);
        }
        #pragma unroll
        for (int i = 0; i < MBLK; ++i)
            #pragma unroll
            for (int j = 0; j < NBLK; ++j)
                acc[i][j] = __builtin_amdgcn_wmma_f32_16x16x32_f16(
                    false, a[i].v, false, b[j].v, (short)0, acc[i][j], false, false);
    }

    #pragma unroll
    for (int i = 0; i < MBLK; ++i) {
        const int rowBase = (mB * MBLK + i) * 16 + hi * 8;
        #pragma unroll
        for (int j = 0; j < NBLK; ++j) {
            const int col = (nB * NBLK + j) * 16 + mr;
            #pragma unroll
            for (int r = 0; r < 8; ++r) {
                const int row = rowBase + r;
                float v = acc[i][j][r];
                if (EPI == 0) {
                    C[(size_t)row * N + col] = v;
                } else if (EPI == 1) {
                    v += bias[col];
                    C[(size_t)row * N + col] = (v > 20.f) ? v : log1pf(__expf(v));
                } else { // EPI == 2: in_proj split epilogue
                    if (col < D_INNER) {
                        C[(size_t)row * D_INNER + col] = v;
                    } else {
                        float s = v / (1.f + __expf(-v));                 // silu(res)
                        out2[(size_t)row * D_INNER + (col - D_INNER)] = (_Float16)s;
                    }
                }
            }
        }
    }
}

// ---------------- depthwise causal conv1d + silu ----------------
__global__ __launch_bounds__(256)
void conv_silu_kernel(const float* __restrict__ xin, const float* __restrict__ cw,
                      const float* __restrict__ cb, float* __restrict__ xc32,
                      _Float16* __restrict__ xc16)
{
    const int idx = blockIdx.x * 256 + threadIdx.x;
    if (idx >= NTOK * D_INNER) return;
    const int d = idx % D_INNER;
    const int l = (idx / D_INNER) % LL;
    const int b = idx / (D_INNER * LL);
    const size_t t = (size_t)b * LL + l;
    float acc = cb[d];
    #pragma unroll
    for (int j = 0; j < D_CONV; ++j) {
        const int ll = l - (D_CONV - 1) + j;
        if (ll >= 0)
            acc += cw[d * D_CONV + j] * xin[((size_t)b * LL + ll) * D_INNER + d];
    }
    const float s = acc / (1.f + __expf(-acc));
    xc32[t * D_INNER + d] = s;
    xc16[t * D_INNER + d] = (_Float16)s;
}

// ---------------- selective scan: lane = (b, d, n), wave = 2 channels x 16 states ----------------
__global__ __launch_bounds__(256)
void scan_kernel(const float* __restrict__ delta, const float* __restrict__ u32,
                 const float* __restrict__ dbc, const _Float16* __restrict__ resg,
                 const float* __restrict__ A_log, const float* __restrict__ Dp,
                 _Float16* __restrict__ yg)
{
    const int g = blockIdx.x * 256 + threadIdx.x;
    if (g >= BB * D_INNER * N_STATE) return;
    const int n = g & 15;
    const int d = (g >> 4) % D_INNER;
    const int b = g / (N_STATE * D_INNER);

    const float Ac = -__expf(A_log[d * N_STATE + n]);   // A = -exp(A_log)
    const float Dv = Dp[d];
    float h = 0.f;
    for (int l = 0; l < LL; ++l) {
        const size_t t = (size_t)b * LL + l;
        const float dv = delta[t * D_INNER + d];
        const float uv = u32[t * D_INNER + d];
        const float Bn = dbc[t * DBC_W + DT_RANK + n];
        const float Cn = dbc[t * DBC_W + DT_RANK + N_STATE + n];
        const float dA = __expf(dv * Ac);
        h = dA * h + dv * Bn * uv;
        float p = h * Cn;
        p += __shfl_xor(p, 1);
        p += __shfl_xor(p, 2);
        p += __shfl_xor(p, 4);
        p += __shfl_xor(p, 8);
        if (n == 0) {
            const float y    = p + uv * Dv;
            const float gate = (float)resg[t * D_INNER + d];   // silu(res), precomputed
            yg[t * D_INNER + d] = (_Float16)(y * gate);
        }
    }
}

// ---------------- launcher ----------------
extern "C" void kernel_launch(void* const* d_in, const int* in_sizes, int n_in,
                              void* d_out, int out_size, void* d_ws, size_t ws_size,
                              hipStream_t stream)
{
    const float* x         = (const float*)d_in[0];
    const float* ln_w      = (const float*)d_in[1];
    const float* ln_b      = (const float*)d_in[2];
    const float* in_proj_w = (const float*)d_in[3];   // (3072, 768)
    const float* conv_w    = (const float*)d_in[4];   // (1536, 1, 4)
    const float* conv_b    = (const float*)d_in[5];
    const float* x_proj_w  = (const float*)d_in[6];   // (80, 1536)
    const float* dt_proj_w = (const float*)d_in[7];   // (1536, 48)
    const float* dt_proj_b = (const float*)d_in[8];
    const float* A_log     = (const float*)d_in[9];   // (1536, 16)
    const float* D_param   = (const float*)d_in[10];
    const float* out_proj_w= (const float*)d_in[11];  // (768, 1536)

    char* ws = (char*)d_ws;
    _Float16* xn16   = (_Float16*)(ws + OFF_XN16);
    _Float16* win16  = (_Float16*)(ws + OFF_WIN16);
    _Float16* wout16 = (_Float16*)(ws + OFF_WOUT16);
    _Float16* wxp16  = (_Float16*)(ws + OFF_WXP16);
    _Float16* wdt16  = (_Float16*)(ws + OFF_WDT16);
    float*    xin32  = (float*)   (ws + OFF_XIN32);
    _Float16* resg16 = (_Float16*)(ws + OFF_RESG16);
    float*    xc32   = (float*)   (ws + OFF_XC32);
    _Float16* xc16   = (_Float16*)(ws + OFF_XC16);
    float*    dbc32  = (float*)   (ws + OFF_DBC32);
    _Float16* dt16   = (_Float16*)(ws + OFF_DT16);
    _Float16* yg16   = (_Float16*)(ws + OFF_YG16);
    float*    delta  = (float*)   (ws + OFF_DELTA);   // aliases xin32 (dead after conv)
    float*    out    = (float*)d_out;

    // 1. layernorm -> f16 activations
    ln_kernel<<<NTOK, 256, 0, stream>>>(x, ln_w, ln_b, xn16);

    // 2. weight conversions to f16 (dt_proj_w padded K 48 -> 64)
    {
        int n = 2 * D_INNER * D_MODEL;
        cvt_kernel<<<(n + 255) / 256, 256, 0, stream>>>(in_proj_w, win16, n);
        n = D_MODEL * D_INNER;
        cvt_kernel<<<(n + 255) / 256, 256, 0, stream>>>(out_proj_w, wout16, n);
        n = DBC_W * D_INNER;
        cvt_kernel<<<(n + 255) / 256, 256, 0, stream>>>(x_proj_w, wxp16, n);
        n = D_INNER * DT_PAD;
        cvt_pad_kernel<<<(n + 255) / 256, 256, 0, stream>>>(dt_proj_w, wdt16,
                                                            D_INNER, DT_RANK, DT_PAD, DT_RANK);
    }

    // 3. in_proj GEMM (4096 x 3072 x 768), 2x4 blocked, split epilogue
    {
        int waves = ((NTOK / 16) / 2) * (((2 * D_INNER) / 16) / 4);   // 128 * 48
        gemm_wmma<2, 4, 2><<<(waves + 7) / 8, 256, 0, stream>>>(xn16, win16, xin32,
                                                                NTOK, 2 * D_INNER, D_MODEL,
                                                                nullptr, resg16);
    }

    // 4. depthwise causal conv + silu -> u (f32 + f16)
    {
        int n = NTOK * D_INNER;
        conv_silu_kernel<<<(n + 255) / 256, 256, 0, stream>>>(xin32, conv_w, conv_b, xc32, xc16);
    }

    // 5. x_proj GEMM (4096 x 80 x 1536), 1x5 blocked -> dbc f32
    {
        int waves = (NTOK / 16) * ((DBC_W / 16) / 5);                 // 256 * 1
        gemm_wmma<1, 5, 0><<<(waves + 7) / 8, 256, 0, stream>>>(xc16, wxp16, dbc32,
                                                                NTOK, DBC_W, D_INNER,
                                                                nullptr, nullptr);
    }

    // 6. dt slice -> f16 padded (4096 x 64)
    {
        int n = NTOK * DT_PAD;
        cvt_pad_kernel<<<(n + 255) / 256, 256, 0, stream>>>(dbc32, dt16,
                                                            NTOK, DBC_W, DT_PAD, DT_RANK);
    }

    // 7. dt_proj GEMM (4096 x 1536 x 64), 2x4 blocked, bias + softplus -> delta (aliases xin32)
    {
        int waves = ((NTOK / 16) / 2) * ((D_INNER / 16) / 4);         // 128 * 24
        gemm_wmma<2, 4, 1><<<(waves + 7) / 8, 256, 0, stream>>>(dt16, wdt16, delta,
                                                                NTOK, D_INNER, DT_PAD,
                                                                dt_proj_b, nullptr);
    }

    // 8. selective scan + gating -> yg f16
    {
        int n = BB * D_INNER * N_STATE;   // 49152 lanes
        scan_kernel<<<(n + 255) / 256, 256, 0, stream>>>(delta, xc32, dbc32, resg16,
                                                         A_log, D_param, yg16);
    }

    // 9. out_proj GEMM (4096 x 768 x 1536), 2x4 blocked -> d_out f32
    {
        int waves = ((NTOK / 16) / 2) * ((D_MODEL / 16) / 4);         // 128 * 12
        gemm_wmma<2, 4, 0><<<(waves + 7) / 8, 256, 0, stream>>>(yg16, wout16, out,
                                                                NTOK, D_MODEL, D_INNER,
                                                                nullptr, nullptr);
    }
}
